// SIN_21801253994515
// MI455X (gfx1250) — matrix operand order repaired and checked by reading, not compiled, not run on hardware
//
#include <hip/hip_runtime.h>

typedef __bf16 bf16;
typedef __attribute__((ext_vector_type(16))) __bf16 v16bf;
typedef __attribute__((ext_vector_type(8)))  __bf16 v8bf;
typedef __attribute__((ext_vector_type(8)))  float  v8f;

#define HD 128

// ---------------------------------------------------------------------------
// One-time weight prep: W[128,128] f32 (k-major) -> Wt[n][k] bf16 (transposed)
// ---------------------------------------------------------------------------
__global__ void __launch_bounds__(256)
wconv_kernel(const float* __restrict__ W, bf16* __restrict__ Wt)
{
  const int i = blockIdx.x * 256 + threadIdx.x;   // 16384 elements
  const int k = i >> 7, n = i & 127;
  Wt[n * HD + k] = (bf16)W[i];
}

// ---------------------------------------------------------------------------
// GEMM: Out[M,128] = act( (A (+ A2)) @ W[128,128] + bias ), bf16 WMMA, f32 acc
// Block = 256 threads (8 waves). Tile: 64 rows x 128 cols, full K=128.
// Weights arrive pre-transposed bf16; fetched via async global->LDS DMA into
// padded rows (272B stride) while threads convert the f32 A tile to bf16.
// ---------------------------------------------------------------------------
__global__ void __launch_bounds__(256)
gemm128_kernel(const float* __restrict__ A, const float* __restrict__ A2,
               const bf16* __restrict__ Wt, const float* __restrict__ bias,
               float* __restrict__ Out, int M, int relu)
{
  __shared__ bf16 sA[64][136];    // +8 pad: row stride 272B -> conflict-free
  __shared__ bf16 sW[128][136];   // sW[n][k], rows filled by async DMA
  const int tid  = threadIdx.x;
  const int row0 = blockIdx.x * 64;

  // Async DMA of 32KB of pre-transposed bf16 weights into padded LDS rows.
  // 16B chunk c: row n = c>>4, sub j = c&15 -> LDS n*272 + j*16, global c*16.
  {
    const unsigned sWbase = (unsigned)(size_t)(void*)&sW[0][0];
    for (int j = 0; j < 8; ++j) {
      const int c = tid + j * 256;
      const unsigned lds = sWbase + (unsigned)((c >> 4) * 272 + (c & 15) * 16);
      const unsigned gof = (unsigned)(c * 16);
      asm volatile("global_load_async_to_lds_b128 %0, %1, %2"
                   :: "v"(lds), "v"(gof), "s"(Wt) : "memory");
    }
  }
  // Overlap: convert A tile f32 -> bf16 while the weight DMA is in flight.
  for (int i = tid; i < 64 * HD; i += 256) {
    int r = i >> 7, c = i & 127;
    int gr = row0 + r;
    float v = 0.f;
    if (gr < M) {
      v = A[(size_t)gr * HD + c];
      if (A2) v += A2[(size_t)gr * HD + c];
    }
    sA[r][c] = (bf16)v;
  }
  asm volatile("s_wait_asynccnt 0x0" ::: "memory");
  __syncthreads();

  const int wave  = tid >> 5, lane = tid & 31;
  const int lrow  = lane & 15, khalf = lane >> 4;
  const int mbase = (wave >> 1) * 16;
  const int nbase = (wave & 1) * 64;

  v8f acc[4];
  for (int t = 0; t < 4; ++t)
    for (int j = 0; j < 8; ++j) acc[t][j] = 0.f;

  for (int kb = 0; kb < HD; kb += 32) {
    // A 16x32 bf16 frag: lanes 0-15 hold K {kb..kb+7, kb+16..kb+23},
    // lanes 16-31 hold K {kb+8..kb+15, kb+24..kb+31} (ISA 16-bit A layout)
    union { v16bf v; v8bf h[2]; } af;
    af.h[0] = *(const v8bf*)&sA[mbase + lrow][kb + khalf * 8];
    af.h[1] = *(const v8bf*)&sA[mbase + lrow][kb + 16 + khalf * 8];
    for (int t = 0; t < 4; ++t) {
      const int col = nbase + t * 16 + lrow;
      // B 32x16 bf16 frag: lane = column; lanes 0-15 K {kb..kb+7, kb+16..kb+23},
      // lanes 16-31 K {kb+8..kb+15, kb+24..kb+31} (same order as A)
      union { v16bf v; v8bf h[2]; } bfr;
      bfr.h[0] = *(const v8bf*)&sW[col][kb + khalf * 8];
      bfr.h[1] = *(const v8bf*)&sW[col][kb + 16 + khalf * 8];
      acc[t] = __builtin_amdgcn_wmma_f32_16x16x32_bf16(
          false, af.v, false, bfr.v, (short)0, acc[t], false, false);
    }
  }

  // C/D layout: VGPR r -> M = r (lanes 0-15) / 8+r (lanes 16-31); N = lane&15
  for (int t = 0; t < 4; ++t) {
    const int col = nbase + t * 16 + lrow;
    const float bv = bias ? bias[col] : 0.f;
    for (int r = 0; r < 8; ++r) {
      const int row = row0 + mbase + khalf * 8 + r;
      if (row < M) {
        float v = acc[t][r] + bv;
        if (relu) v = fmaxf(v, 0.f);
        Out[(size_t)row * HD + col] = v;
      }
    }
  }
}

// ---------------------------------------------------------------------------
// m[e] = relu(Tsrc[i0[e]] + Tattr[sh[e]] + bias); store + column sum/sumsq
// ---------------------------------------------------------------------------
__global__ void __launch_bounds__(128)
msg_build_kernel(const float* __restrict__ Tsrc, const float* __restrict__ Tattr,
                 const float* __restrict__ bias, const int* __restrict__ i0,
                 const int* __restrict__ sh, float* __restrict__ Mbuf,
                 float* __restrict__ stats, int E)
{
  const int c = threadIdx.x;
  const float b = bias[c];
  float s = 0.f, s2 = 0.f;
  for (int e = blockIdx.x; e < E; e += gridDim.x) {
    const int src = i0[e], at = sh[e];
    float m = Tsrc[(size_t)src * HD + c] + Tattr[(size_t)at * HD + c] + b;
    m = fmaxf(m, 0.f);
    Mbuf[(size_t)e * HD + c] = m;
    s += m; s2 += m * m;
  }
  atomicAdd(&stats[c], s);
  atomicAdd(&stats[HD + c], s2);
}

__global__ void bn_finalize_kernel(const float* __restrict__ stats,
                                   const float* __restrict__ gamma,
                                   const float* __restrict__ beta,
                                   float* __restrict__ bnp, float count)
{
  const int c = threadIdx.x;
  const float mean = stats[c] / count;
  float var = stats[HD + c] / count - mean * mean;
  const float sc = gamma[c] * rsqrtf(var + 1e-5f);
  bnp[c]      = sc;
  bnp[HD + c] = beta[c] - mean * sc;
}

__global__ void __launch_bounds__(128)
msg_scatter_kernel(const float* __restrict__ Mbuf, const float* __restrict__ bnp,
                   const int* __restrict__ i1, float* __restrict__ Agg, int E)
{
  const int c = threadIdx.x;
  const float sc = bnp[c], sh = bnp[HD + c];
  for (int e = blockIdx.x; e < E; e += gridDim.x) {
    const int dst = i1[e];
    atomicAdd(&Agg[(size_t)dst * HD + c], Mbuf[(size_t)e * HD + c] * sc + sh);
  }
}

__global__ void __launch_bounds__(128)
colstats_kernel(const float* __restrict__ X, float* __restrict__ stats, int N)
{
  const int c = threadIdx.x;
  float s = 0.f, s2 = 0.f;
  for (int r = blockIdx.x; r < N; r += gridDim.x) {
    const float v = X[(size_t)r * HD + c];
    s += v; s2 += v * v;
  }
  atomicAdd(&stats[c], s);
  atomicAdd(&stats[HD + c], s2);
}

__global__ void bn_apply_kernel(const float* __restrict__ X,
                                const float* __restrict__ bnp,
                                float* __restrict__ Y, size_t n)
{
  size_t i = (size_t)blockIdx.x * blockDim.x + threadIdx.x;
  if (i < n) {
    const int c = (int)(i & 127);
    Y[i] = X[i] * bnp[c] + bnp[HD + c];
  }
}

__global__ void zero_kernel(float* __restrict__ p, size_t n)
{
  size_t i = (size_t)blockIdx.x * blockDim.x + threadIdx.x;
  const size_t stride = (size_t)gridDim.x * blockDim.x;
  for (; i < n; i += stride) p[i] = 0.f;
}

__global__ void __launch_bounds__(128)
pool_kernel(const float* __restrict__ X, const int* __restrict__ batch,
            float* __restrict__ pool, float* __restrict__ cnt, int N)
{
  const int c = threadIdx.x;
  for (int r = blockIdx.x; r < N; r += gridDim.x) {
    const int b = batch[r];
    atomicAdd(&pool[(size_t)b * HD + c], X[(size_t)r * HD + c]);
    if (c == 0) atomicAdd(&cnt[b], 1.f);
  }
}

__global__ void combine_pool_kernel(const float* __restrict__ pool,
                                    const float* __restrict__ cnt,
                                    float* __restrict__ g)
{
  const int idx = blockIdx.x * blockDim.x + threadIdx.x;
  if (idx >= 64 * HD) return;
  const int b = idx >> 7;
  float v = 0.f;
  for (int d = 0; d < 3; ++d)
    v += pool[d * 64 * HD + idx] / fmaxf(cnt[d * 64 + b], 1.f);
  g[idx] = v;
}

__global__ void __launch_bounds__(32)
head_final_kernel(const float* __restrict__ Hm, const float* __restrict__ W2,
                  const float* __restrict__ b2, float* __restrict__ out)
{
  const int b = blockIdx.x, lane = threadIdx.x;
  __shared__ float logits[10];
  if (lane < 10) {
    float acc = b2[lane];
    for (int k = 0; k < HD; ++k) acc += Hm[(size_t)b * HD + k] * W2[k * 10 + lane];
    logits[lane] = acc;
  }
  __syncthreads();
  if (lane == 0) {
    float mx = logits[0];
    for (int i = 1; i < 10; ++i) mx = fmaxf(mx, logits[i]);
    float se = 0.f;
    for (int i = 0; i < 10; ++i) se += expf(logits[i] - mx);
    const float lse = mx + logf(se);
    for (int i = 0; i < 10; ++i) out[b * 10 + i] = logits[i] - lse;
  }
}

// ---------------------------------------------------------------------------
extern "C" void kernel_launch(void* const* d_in, const int* in_sizes, int n_in,
                              void* d_out, int out_size, void* d_ws, size_t ws_size,
                              hipStream_t stream)
{
  (void)in_sizes; (void)out_size; (void)ws_size;
  const int N0 = 100000, N1 = 200000, N2 = 80000, Bn = 64;
  const int EU0 = 300000, EU1 = 300000, ED1 = 400000, ED2 = 240000;

  const float* x0 = (const float*)d_in[0];
  const float* x1 = (const float*)d_in[1];
  const float* x2 = (const float*)d_in[2];
  const int* up_index0   = (const int*)d_in[3];
  const int* shared_cob0 = (const int*)d_in[4];
  const int* up_index1   = (const int*)d_in[5];
  const int* shared_cob1 = (const int*)d_in[6];
  const int* down_index1 = (const int*)d_in[7];
  const int* shared_bnd1 = (const int*)d_in[8];
  const int* down_index2 = (const int*)d_in[9];
  const int* shared_bnd2 = (const int*)d_in[10];
  const int* batch0 = (const int*)d_in[11];
  const int* batch1 = (const int*)d_in[12];
  const int* batch2 = (const int*)d_in[13];

  // params: jax pytree sorted-key flatten. Per layer (x3):
  //   down{W,b,be,g}, up{W,b,be,g}, upd{W1,W2,b1,b2,be,g}; then lin1_W/b, lin2_W/b
  static const int leaf_sz[46] = {
    32768,128,128,128, 32768,128,128,128, 16384,16384,128,128,128,128,
    32768,128,128,128, 32768,128,128,128, 16384,16384,128,128,128,128,
    32768,128,128,128, 32768,128,128,128, 16384,16384,128,128,128,128,
    16384,128,1280,10 };
  const float* P[46];
  if (n_in >= 14 + 46) {
    for (int i = 0; i < 46; ++i) P[i] = (const float*)d_in[14 + i];
  } else {
    const float* base = (const float*)d_in[14];
    size_t o = 0;
    for (int i = 0; i < 46; ++i) { P[i] = base + o; o += leaf_sz[i]; }
  }

  // workspace carve-up (floats)
  float* ws = (float*)d_ws;
  size_t off = 0;
  auto take = [&](size_t n) { float* p = ws + off; off += (n + 63) & ~(size_t)63; return p; };
  const size_t NT = (size_t)(N0 + N1 + N2) * HD;
  float* fA    = take(NT);
  float* fB    = take(NT);
  float* Tsrc  = take((size_t)N1 * HD);   // reused as h2 in update phase
  float* Tattr = take((size_t)N1 * HD);
  float* Mbuf  = take((size_t)ED1 * HD);  // reused as h1 in update phase
  float* agg0  = take((size_t)N0 * HD);
  float* agg1  = take((size_t)N1 * HD);
  float* agg2  = take((size_t)N2 * HD);
  float* stats = take(256);
  float* bnp   = take(256);
  float* pool  = take((size_t)3 * Bn * HD);
  float* cnt   = take((size_t)3 * Bn);
  float* gbuf  = take((size_t)Bn * HD);
  float* hbuf  = take((size_t)Bn * HD);
  bf16*  wbf   = (bf16*)take((size_t)19 * 16384 / 2);  // 19 transposed bf16 mats
  float* h1buf = Mbuf;
  float* h2buf = Tsrc;

  auto zero = [&](float* p, size_t n) {
    int g = (int)((n + 255) / 256);
    zero_kernel<<<g, 256, 0, stream>>>(p, n);
  };
  auto wconv = [&](const float* W, bf16* Wt) {
    wconv_kernel<<<64, 256, 0, stream>>>(W, Wt);
  };
  auto gemm = [&](const float* A, const float* A2, const bf16* Wt,
                  const float* bias, float* Out, int M, int relu) {
    gemm128_kernel<<<(M + 63) / 64, 256, 0, stream>>>(A, A2, Wt, bias, Out, M, relu);
  };

  // Pre-transpose / bf16-convert all weight matrices once per launch.
  // Per layer l: [0]=up.W top, [1]=up.W bot, [2]=down.W top, [3]=down.W bot,
  //              [4]=upd.W1, [5]=upd.W2 ; slot 18 = lin1_W.
  for (int l = 0; l < 3; ++l) {
    const int lb = l * 14;
    bf16* base = wbf + (size_t)l * 6 * 16384;
    wconv(P[lb + 4],         base + 0 * 16384);   // up.W rows 0..127
    wconv(P[lb + 4] + 16384, base + 1 * 16384);   // up.W rows 128..255
    wconv(P[lb + 0],         base + 2 * 16384);   // down.W top
    wconv(P[lb + 0] + 16384, base + 3 * 16384);   // down.W bot
    wconv(P[lb + 8],         base + 4 * 16384);   // upd.W1
    wconv(P[lb + 9],         base + 5 * 16384);   // upd.W2
  }
  wconv(P[42], wbf + (size_t)18 * 16384);         // lin1_W

  // factored concat-GEMM aggregation: Tsrc = src@W_top, Tattr = attr@W_bot,
  // then fused gather/ReLU/stats, BN finalize, normalize+scatter-add.
  auto agg = [&](const float* srcX, int Nsrc, const float* attrX, int Nattr,
                 const bf16* WtTop, const bf16* WtBot, const float* bm,
                 const float* gm, const float* bem, const int* idx,
                 const int* shv, int E, float* aggOut) {
    gemm(srcX,  nullptr, WtTop, nullptr, Tsrc,  Nsrc,  0);
    gemm(attrX, nullptr, WtBot, nullptr, Tattr, Nattr, 0);
    zero(stats, 256);
    msg_build_kernel<<<2048, 128, 0, stream>>>(Tsrc, Tattr, bm, idx, shv, Mbuf, stats, E);
    bn_finalize_kernel<<<1, 128, 0, stream>>>(stats, gm, bem, bnp, (float)E);
    msg_scatter_kernel<<<2048, 128, 0, stream>>>(Mbuf, bnp, idx + E, aggOut, E);
  };

  const float* c0 = x0; const float* c1 = x1; const float* c2 = x2;
  for (int l = 0; l < 3; ++l) {
    const int lb = l * 14;
    const float* db_   = P[lb + 1];
    const float* dbe   = P[lb + 2];  const float* dg   = P[lb + 3];
    const float* ub_   = P[lb + 5];
    const float* ube_m = P[lb + 6];  const float* ug_m = P[lb + 7];
    const float* b1    = P[lb + 10]; const float* b2   = P[lb + 11];
    const float* ube   = P[lb + 12]; const float* ug   = P[lb + 13];
    const bf16* wB = wbf + (size_t)l * 6 * 16384;
    const bf16* uWt_top = wB + 0 * 16384, *uWt_bot = wB + 1 * 16384;
    const bf16* dWt_top = wB + 2 * 16384, *dWt_bot = wB + 3 * 16384;
    const bf16* W1t = wB + 4 * 16384,     *W2t = wB + 5 * 16384;

    float* nb = (l % 2 == 0) ? fA : fB;
    float* n0 = nb;
    float* n1 = n0 + (size_t)N0 * HD;
    float* n2 = n1 + (size_t)N1 * HD;

    zero(agg0, (size_t)N0 * HD);
    zero(agg1, (size_t)N1 * HD);
    zero(agg2, (size_t)N2 * HD);

    agg(c0, N0, c1, N1, uWt_top, uWt_bot, ub_, ug_m, ube_m, up_index0,   shared_cob0, EU0, agg0);
    agg(c1, N1, c2, N2, uWt_top, uWt_bot, ub_, ug_m, ube_m, up_index1,   shared_cob1, EU1, agg1);
    agg(c1, N1, c0, N0, dWt_top, dWt_bot, db_, dg,   dbe,   down_index1, shared_bnd1, ED1, agg1);
    agg(c2, N2, c1, N1, dWt_top, dWt_bot, db_, dg,   dbe,   down_index2, shared_bnd2, ED2, agg2);

    auto upd = [&](const float* cx, const float* ag, int Nd, float* outx) {
      gemm(cx, ag, W1t, b1, h1buf, Nd, 1);
      gemm(h1buf, nullptr, W2t, b2, h2buf, Nd, 1);
      zero(stats, 256);
      colstats_kernel<<<2048, 128, 0, stream>>>(h2buf, stats, Nd);
      bn_finalize_kernel<<<1, 128, 0, stream>>>(stats, ug, ube, bnp, (float)Nd);
      const size_t tot = (size_t)Nd * HD;
      bn_apply_kernel<<<(int)((tot + 255) / 256), 256, 0, stream>>>(h2buf, bnp, outx, tot);
    };
    upd(c0, agg0, N0, n0);
    upd(c1, agg1, N1, n1);
    upd(c2, agg2, N2, n2);
    c0 = n0; c1 = n1; c2 = n2;
  }

  zero(pool, (size_t)3 * Bn * HD);
  zero(cnt,  (size_t)3 * Bn);
  pool_kernel<<<2048, 128, 0, stream>>>(c0, batch0, pool,               cnt,          N0);
  pool_kernel<<<2048, 128, 0, stream>>>(c1, batch1, pool + Bn * HD,     cnt + Bn,     N1);
  pool_kernel<<<2048, 128, 0, stream>>>(c2, batch2, pool + 2 * Bn * HD, cnt + 2 * Bn, N2);
  combine_pool_kernel<<<(Bn * HD + 255) / 256, 256, 0, stream>>>(pool, cnt, gbuf);
  gemm(gbuf, nullptr, wbf + (size_t)18 * 16384, P[43], hbuf, Bn, 1);
  head_final_kernel<<<Bn, 32, 0, stream>>>(hbuf, P[44], P[45], (float*)d_out);
}
